// AttentionPool2d_46016279610060
// MI455X (gfx1250) — matrix operand order repaired and checked by reading, not compiled, not run on hardware
//
#include <hip/hip_runtime.h>
#include <hip/hip_bf16.h>

typedef __attribute__((ext_vector_type(16))) _Float16 v16h;
typedef __attribute__((ext_vector_type(8)))  _Float16 v8h;
typedef __attribute__((ext_vector_type(8)))  float    v8f;

#define B_   2
#define H_   16
#define E_   1024
#define HD_  64
#define SP_  14
#define NM_  100
#define L_   197          // SP*SP + 1
#define R_   (L_ * B_)    // 394 rows (token-major, batch-minor)
#define RP_  400          // padded to 25 * 16
#define MT_  25
#define NQ_  16           // 16 quads of 64 columns = 1024

// ---------------------------------------------------------------------------
// K0: build token matrix XS[l*B+b][e] = x-token + pos_emb, mean token at l=0.
//     Writes f16 XS (padded rows zeroed) and f32 copy of row l=0 per batch.
// ---------------------------------------------------------------------------
__global__ void k_build_xs(const float* __restrict__ x, const float* __restrict__ pos,
                           _Float16* __restrict__ xs16, float* __restrict__ xs0) {
    int idx = blockIdx.x * blockDim.x + threadIdx.x;     // row*1024 + e
    if (idx >= RP_ * E_) return;
    int row = idx >> 10, e = idx & 1023;
    if (row >= R_) { xs16[idx] = (_Float16)0.0f; return; }
    int l = row / B_, b = row % B_;
    float v;
    if (l == 0) {
        const float* xp = x + ((size_t)b * E_ + e) * (SP_ * SP_);
        float s = 0.0f;
        for (int p = 0; p < SP_ * SP_; ++p) s += xp[p];
        v = s * (1.0f / (SP_ * SP_)) + pos[e];
        xs0[b * E_ + e] = v;
    } else {
        v = x[((size_t)b * E_ + e) * (SP_ * SP_) + (l - 1)] + pos[(size_t)l * E_ + e];
    }
    xs16[idx] = (_Float16)v;
}

// ---------------------------------------------------------------------------
// K1: f32 -> f16 weight conversion (row-major preserved; B-frag reads it as W^T)
// ---------------------------------------------------------------------------
__global__ void k_f32_to_f16(const float* __restrict__ in, _Float16* __restrict__ out, int n) {
    int i = blockIdx.x * blockDim.x + threadIdx.x;
    if (i < n) out[i] = (_Float16)in[i];
}

// ---------------------------------------------------------------------------
// K2: q0[b][e] = (xs0[b] . q_w[e]) + q_b[e], scaled by hd^-0.5 = 0.125
// ---------------------------------------------------------------------------
__global__ void k_q0(const float* __restrict__ xs0, const float* __restrict__ qw,
                     const float* __restrict__ qb, float* __restrict__ q0) {
    int idx = blockIdx.x * blockDim.x + threadIdx.x;
    if (idx >= B_ * E_) return;
    int b = idx >> 10, e = idx & 1023;
    const float* xr = xs0 + b * E_;
    const float* wr = qw + (size_t)e * E_;
    float s = 0.0f;
    for (int i = 0; i < E_; ++i) s += xr[i] * wr[i];
    q0[idx] = (s + qb[e]) * 0.125f;
}

// ---------------------------------------------------------------------------
// K3: WMMA projection  out[RP_ x 1024] = XS_f16 @ W_f16^T + bias
//     One wave (32 threads) per 16x64 output strip.
//     A frag: lane holds row (lane&15); lanes 0-15: K {0..15},
//             lanes 16-31: K {16..31}  -> two v8h loads.
//     B frag: lane holds col (lane&15), K = 16*(lane>>4)..+15; B = W^T so this
//             is one contiguous v16h load from row-major W.
//     D frag: VGPR r -> row r + 8*(lane>>4), col lane&15.
// ---------------------------------------------------------------------------
__global__ void k_proj_wmma(const _Float16* __restrict__ XS, const _Float16* __restrict__ W,
                            const float* __restrict__ bias, float* __restrict__ out) {
    const int m0   = blockIdx.x * 16;
    const int n0   = blockIdx.y * 64;
    const int lane = threadIdx.x;
    const int col  = lane & 15;
    const int half = lane >> 4;

    v8f acc0 = {}, acc1 = {}, acc2 = {}, acc3 = {};
    const _Float16* arow = XS + (size_t)(m0 + col) * E_;

    for (int k0 = 0; k0 < E_; k0 += 32) {
        v8h alo = *(const v8h*)(arow + k0 + half * 16);
        v8h ahi = *(const v8h*)(arow + k0 + half * 16 + 8);
        v16h a;
#pragma unroll
        for (int j = 0; j < 8; ++j) { a[j] = alo[j]; a[8 + j] = ahi[j]; }

        const _Float16* wk = W + k0 + half * 16;
        v16h b0 = *(const v16h*)(wk + (size_t)(n0 +  0 + col) * E_);
        v16h b1 = *(const v16h*)(wk + (size_t)(n0 + 16 + col) * E_);
        v16h b2 = *(const v16h*)(wk + (size_t)(n0 + 32 + col) * E_);
        v16h b3 = *(const v16h*)(wk + (size_t)(n0 + 48 + col) * E_);

        acc0 = __builtin_amdgcn_wmma_f32_16x16x32_f16(false, a, false, b0, (short)0, acc0, false, false);
        acc1 = __builtin_amdgcn_wmma_f32_16x16x32_f16(false, a, false, b1, (short)0, acc1, false, false);
        acc2 = __builtin_amdgcn_wmma_f32_16x16x32_f16(false, a, false, b2, (short)0, acc2, false, false);
        acc3 = __builtin_amdgcn_wmma_f32_16x16x32_f16(false, a, false, b3, (short)0, acc3, false, false);
    }

#pragma unroll
    for (int r = 0; r < 8; ++r) {
        int rg = m0 + r + half * 8;
        if (rg < R_) {
            float* orow = out + (size_t)rg * E_;
            int c0 = n0 + col;
            orow[c0     ] = acc0[r] + bias[c0     ];
            orow[c0 + 16] = acc1[r] + bias[c0 + 16];
            orow[c0 + 32] = acc2[r] + bias[c0 + 32];
            orow[c0 + 48] = acc3[r] + bias[c0 + 48];
        }
    }
}

// ---------------------------------------------------------------------------
// K4: masks[b][n][t]: t==0 -> 1.0 else sigmoid(mask_feature[b][n][8i][8j])
// ---------------------------------------------------------------------------
__global__ void k_masks(const float* __restrict__ mf, float* __restrict__ m) {
    int idx = blockIdx.x * blockDim.x + threadIdx.x;
    if (idx >= B_ * NM_ * L_) return;
    int t = idx % L_;
    int rest = idx / L_;
    int n = rest % NM_, b = rest / NM_;
    float v;
    if (t == 0) {
        v = 1.0f;
    } else {
        int p = t - 1, i = p / SP_, j = p % SP_;
        float z = mf[(((size_t)b * NM_ + n) * (SP_ * 8) + i * 8) * (SP_ * 8) + j * 8];
        v = 1.0f / (1.0f + __expf(-z));
    }
    m[idx] = v;
}

// ---------------------------------------------------------------------------
// K5: scores[b][h][t] = q0[b][h*64..] . Kmat[t*B+b][h*64..]
// ---------------------------------------------------------------------------
__global__ void k_scores(const float* __restrict__ q0, const float* __restrict__ K,
                         float* __restrict__ scores) {
    int idx = blockIdx.x * blockDim.x + threadIdx.x;
    if (idx >= B_ * H_ * L_) return;
    int t = idx % L_;
    int h = (idx / L_) % H_;
    int b = idx / (H_ * L_);
    const float* qp = q0 + (size_t)b * E_ + h * HD_;
    const float* kp = K + ((size_t)t * B_ + b) * E_ + h * HD_;
    float s = 0.0f;
    for (int d = 0; d < HD_; ++d) s += qp[d] * kp[d];
    scores[idx] = s;
}

// ---------------------------------------------------------------------------
// K6: wacc[b][h][t] = sum_n softmax_t( scores[b][h][t] * masks[b][n][t] )
//     one workgroup per (b,h); 8 waves, each wave owns masks n = wave+8k;
//     wave32 shfl_xor reductions for max and sum; LDS float atomics to
//     accumulate over masks.
// ---------------------------------------------------------------------------
__global__ void k_softmax_acc(const float* __restrict__ scores, const float* __restrict__ masks,
                              float* __restrict__ wacc) {
    __shared__ float s_sc[L_];
    __shared__ float s_w[L_];
    const int bh  = blockIdx.x;          // b*H_ + h
    const int b   = bh / H_;
    const int tid = threadIdx.x;

    for (int t = tid; t < L_; t += blockDim.x) {
        s_sc[t] = scores[(size_t)bh * L_ + t];
        s_w[t]  = 0.0f;
    }
    __syncthreads();

    const int wave = tid >> 5, lane = tid & 31;
    for (int n = wave; n < NM_; n += 8) {
        const float* mp = masks + ((size_t)b * NM_ + n) * L_;
        float v[7];
        float mx = -1e30f;
#pragma unroll
        for (int i = 0; i < 7; ++i) {
            int t = lane + i * 32;
            float vv = (t < L_) ? s_sc[t] * mp[t] : -1e30f;
            v[i] = vv;
            mx = fmaxf(mx, vv);
        }
        for (int off = 16; off > 0; off >>= 1) mx = fmaxf(mx, __shfl_xor(mx, off, 32));
        float sum = 0.0f;
#pragma unroll
        for (int i = 0; i < 7; ++i) {
            int t = lane + i * 32;
            float e = (t < L_) ? __expf(v[i] - mx) : 0.0f;
            v[i] = e;
            sum += e;
        }
        for (int off = 16; off > 0; off >>= 1) sum += __shfl_xor(sum, off, 32);
        float inv = 1.0f / sum;
#pragma unroll
        for (int i = 0; i < 7; ++i) {
            int t = lane + i * 32;
            if (t < L_) atomicAdd(&s_w[t], v[i] * inv);
        }
    }
    __syncthreads();
    for (int t = tid; t < L_; t += blockDim.x) wacc[(size_t)bh * L_ + t] = s_w[t];
}

// ---------------------------------------------------------------------------
// K7: attn0[b][e] = sum_t wacc[b][h(e)][t] * Vmat[t*B+b][e]
// ---------------------------------------------------------------------------
__global__ void k_attn0(const float* __restrict__ wacc, const float* __restrict__ V,
                        float* __restrict__ attn0) {
    int idx = blockIdx.x * blockDim.x + threadIdx.x;
    if (idx >= B_ * E_) return;
    int b = idx >> 10, e = idx & 1023, h = e / HD_;
    const float* wp = wacc + ((size_t)b * H_ + h) * L_;
    float s = 0.0f;
    for (int t = 0; t < L_; ++t) s += wp[t] * V[((size_t)t * B_ + b) * E_ + e];
    attn0[idx] = s;
}

// ---------------------------------------------------------------------------
// K8: out[b][o] = attn0[b] . c_w[o] + c_b[o]
// ---------------------------------------------------------------------------
__global__ void k_out(const float* __restrict__ attn0, const float* __restrict__ cw,
                      const float* __restrict__ cb, float* __restrict__ out) {
    int idx = blockIdx.x * blockDim.x + threadIdx.x;
    if (idx >= B_ * E_) return;
    int b = idx >> 10, o = idx & 1023;
    const float* ap = attn0 + (size_t)b * E_;
    const float* wr = cw + (size_t)o * E_;
    float s = 0.0f;
    for (int i = 0; i < E_; ++i) s += ap[i] * wr[i];
    out[idx] = s + cb[o];
}

// ---------------------------------------------------------------------------
extern "C" void kernel_launch(void* const* d_in, const int* in_sizes, int n_in,
                              void* d_out, int out_size, void* d_ws, size_t ws_size,
                              hipStream_t stream) {
    const float* x    = (const float*)d_in[0];
    const float* mf   = (const float*)d_in[1];
    const float* pos  = (const float*)d_in[2];
    const float* q_w  = (const float*)d_in[3];
    const float* q_b  = (const float*)d_in[4];
    const float* k_w  = (const float*)d_in[5];
    const float* k_b  = (const float*)d_in[6];
    const float* v_w  = (const float*)d_in[7];
    const float* v_b  = (const float*)d_in[8];
    const float* c_w  = (const float*)d_in[9];
    const float* c_b  = (const float*)d_in[10];
    float* out = (float*)d_out;

    // workspace carve-up (256B aligned), ~8.5 MB total
    char* base = (char*)d_ws;
    size_t off = 0;
    auto carve = [&](size_t bytes) -> char* {
        char* p = base + off;
        off = (off + bytes + 255) & ~(size_t)255;
        return p;
    };
    _Float16* xs16  = (_Float16*)carve((size_t)RP_ * E_ * 2);
    _Float16* kw16  = (_Float16*)carve((size_t)E_ * E_ * 2);
    _Float16* vw16  = (_Float16*)carve((size_t)E_ * E_ * 2);
    float*    Kmat  = (float*)   carve((size_t)RP_ * E_ * 4);
    float*    Vmat  = (float*)   carve((size_t)RP_ * E_ * 4);
    float*    xs0   = (float*)   carve((size_t)B_ * E_ * 4);
    float*    q0    = (float*)   carve((size_t)B_ * E_ * 4);
    float*    masks = (float*)   carve((size_t)B_ * NM_ * L_ * 4);
    float*    scr   = (float*)   carve((size_t)B_ * H_ * L_ * 4);
    float*    wacc  = (float*)   carve((size_t)B_ * H_ * L_ * 4);
    float*    attn0 = (float*)   carve((size_t)B_ * E_ * 4);
    (void)ws_size; (void)in_sizes; (void)n_in; (void)out_size;

    k_build_xs<<<(RP_ * E_ + 255) / 256, 256, 0, stream>>>(x, pos, xs16, xs0);
    k_f32_to_f16<<<(E_ * E_ + 255) / 256, 256, 0, stream>>>(k_w, kw16, E_ * E_);
    k_f32_to_f16<<<(E_ * E_ + 255) / 256, 256, 0, stream>>>(v_w, vw16, E_ * E_);
    k_q0<<<(B_ * E_ + 255) / 256, 256, 0, stream>>>(xs0, q_w, q_b, q0);

    dim3 gproj(MT_, NQ_);
    k_proj_wmma<<<gproj, 32, 0, stream>>>(xs16, kw16, k_b, Kmat);
    k_proj_wmma<<<gproj, 32, 0, stream>>>(xs16, vw16, v_b, Vmat);

    k_masks<<<(B_ * NM_ * L_ + 255) / 256, 256, 0, stream>>>(mf, masks);
    k_scores<<<(B_ * H_ * L_ + 255) / 256, 256, 0, stream>>>(q0, Kmat, scr);
    k_softmax_acc<<<B_ * H_, 256, 0, stream>>>(scr, masks, wacc);
    k_attn0<<<(B_ * E_ + 255) / 256, 256, 0, stream>>>(wacc, Vmat, attn0);
    k_out<<<(B_ * E_ + 255) / 256, 256, 0, stream>>>(attn0, c_w, c_b, out);
}
